// OffsetAttention2_65352222376316
// MI455X (gfx1250) — matrix operand order repaired and checked by reading, not compiled, not run on hardware
//
#include <hip/hip_runtime.h>

typedef __attribute__((ext_vector_type(16))) _Float16 v16h;
typedef __attribute__((ext_vector_type(8)))  float    v8f;
typedef __attribute__((ext_vector_type(2)))  float    v2f;
typedef __attribute__((ext_vector_type(4)))  unsigned int u32x4;

union V16 { v16h v; u32x4 q[2]; };

#define NB    16
#define CIN   64
#define CC    128
#define CQ    32
#define NN    2048
#define BN_EPS 1e-5f

// ---------------- stem: h = relu(bn1(conv1_w @ x)) ----------------
__global__ __launch_bounds__(256) void k_stem(
    const float* __restrict__ x, const float* __restrict__ w,
    const float* __restrict__ g, const float* __restrict__ be,
    const float* __restrict__ mu, const float* __restrict__ var,
    float* __restrict__ h)
{
  __shared__ float xt[CIN][16];
  const int b = blockIdx.y, n0 = blockIdx.x * 16, tid = threadIdx.x;
#pragma unroll
  for (int i = 0; i < 4; ++i) {
    int e = tid + (i << 8);
    xt[e >> 4][e & 15] = x[((size_t)b * CIN + (e >> 4)) * NN + n0 + (e & 15)];
  }
  __syncthreads();
#pragma unroll
  for (int i = 0; i < 8; ++i) {
    int e = tid + (i << 8);
    int c = e >> 4, nl = e & 15;
    float acc = 0.f;
#pragma unroll
    for (int k = 0; k < CIN; ++k) acc += w[c * CIN + k] * xt[k][nl];
    float sc = g[c] * rsqrtf(var[c] + BN_EPS);
    float vv = acc * sc + (be[c] - mu[c] * sc);
    h[((size_t)b * CC + c) * NN + n0 + nl] = fmaxf(vv, 0.f);
  }
}

// ------------- qkv: qkT32[b,n,q] (f32), xv[b,c,n] = v_w@h + v_b (f16) -------------
__global__ __launch_bounds__(256) void k_qkv(
    const float* __restrict__ h, const float* __restrict__ qkw,
    const float* __restrict__ vw, const float* __restrict__ vb,
    float* __restrict__ qkT32, _Float16* __restrict__ xv)
{
  __shared__ float ht[CC][16];
  const int b = blockIdx.y, n0 = blockIdx.x * 16, tid = threadIdx.x;
#pragma unroll
  for (int i = 0; i < 8; ++i) {
    int e = tid + (i << 8);
    ht[e >> 4][e & 15] = h[((size_t)b * CC + (e >> 4)) * NN + n0 + (e & 15)];
  }
  __syncthreads();
  // qk: 32 x 16 outputs (q fastest for coalesced [n,q] stores)
#pragma unroll
  for (int i = 0; i < 2; ++i) {
    int e = tid + (i << 8);
    int q = e & 31, nl = e >> 5;
    float acc = 0.f;
#pragma unroll
    for (int c = 0; c < CC; ++c) acc += qkw[q * CC + c] * ht[c][nl];
    qkT32[((size_t)b * NN + n0 + nl) * CQ + q] = acc;
  }
  // xv: 128 x 16 outputs
#pragma unroll
  for (int i = 0; i < 8; ++i) {
    int e = tid + (i << 8);
    int o = e >> 4, nl = e & 15;
    float acc = vb[o];
#pragma unroll
    for (int c = 0; c < CC; ++c) acc += vw[o * CC + c] * ht[c][nl];
    xv[((size_t)b * CC + o) * NN + n0 + nl] = (_Float16)acc;
  }
}

// ------- energy[b,n,m] = sum_q qk[q,n]*qk[q,m] : exact f32 WMMA 16x16x4, 8 chained -------
__global__ __launch_bounds__(256) void k_energy(
    const float* __restrict__ qkT32, float* __restrict__ energy)
{
  const int tid = threadIdx.x;
  const int lane = tid & 31;
  const int wid = (blockIdx.x << 3) + (tid >> 5);   // 32768 wave jobs
  const int b = wid >> 11;
  const int rem = wid & 2047;
  const int n0 = (rem >> 4) * 16;
  const int m0 = (rem & 15) * 128;
  const int hl = lane >> 4, ml = lane & 15;

  // A fragment (f32 16x4 per step): lane row = n0+ml, step s covers K = 4s + 2*hl + {0,1}
  const float* arow = qkT32 + ((size_t)b * NN + n0 + ml) * CQ + 2 * hl;
  v2f a[8];
#pragma unroll
  for (int s = 0; s < 8; ++s) a[s] = *(const v2f*)(arow + 4 * s);

#pragma unroll
  for (int mt = 0; mt < 8; ++mt) {
    int col = m0 + mt * 16 + ml;
    const float* brow = qkT32 + ((size_t)b * NN + col) * CQ + 2 * hl;
    v8f acc = {};
#pragma unroll
    for (int s = 0; s < 8; ++s) {
      v2f bb = *(const v2f*)(brow + 4 * s);
      acc = __builtin_amdgcn_wmma_f32_16x16x4_f32(
          false, a[s], false, bb, (short)0, acc, false, false);
    }
#pragma unroll
    for (int r = 0; r < 8; ++r) {
      int row = n0 + r + hl * 8;
      energy[((size_t)b * NN + row) * NN + col] = acc[r];
    }
  }
}

// ------------- per-row {max, 1/sum(exp)} packed as float2 -------------
__global__ __launch_bounds__(256) void k_rowstats(
    const float* __restrict__ energy, v2f* __restrict__ rstat)
{
  __shared__ float red[256];
  const int tid = threadIdx.x;
  const size_t base = (size_t)blockIdx.x << 11;
  float v[8];
  float m = -3.402823e38f;
#pragma unroll
  for (int i = 0; i < 8; ++i) { v[i] = energy[base + tid + (i << 8)]; m = fmaxf(m, v[i]); }
  red[tid] = m; __syncthreads();
  for (int s = 128; s > 0; s >>= 1) { if (tid < s) red[tid] = fmaxf(red[tid], red[tid + s]); __syncthreads(); }
  m = red[0]; __syncthreads();
  float sum = 0.f;
#pragma unroll
  for (int i = 0; i < 8; ++i) sum += __expf(v[i] - m);
  red[tid] = sum; __syncthreads();
  for (int s = 128; s > 0; s >>= 1) { if (tid < s) red[tid] += red[tid + s]; __syncthreads(); }
  if (tid == 0) { v2f r; r.x = m; r.y = 1.f / red[0]; rstat[blockIdx.x] = r; }
}

// ------------- column-sum partials (deterministic two-level), with prefetch -------------
__global__ __launch_bounds__(256) void k_colpart(
    const float* __restrict__ energy, const v2f* __restrict__ rstat,
    float* __restrict__ partial)
{
  const int tid = threadIdx.x;
  const int b  = blockIdx.x >> 6;
  const int sub = blockIdx.x & 63;
  const int m  = (sub >> 3) * 256 + tid;
  const int nb = (sub & 7) * 256;
  float acc = 0.f;
#pragma unroll 4
  for (int i = 0; i < 256; ++i) {
    int n = nb + i;
    if (n + 32 < NN)
      __builtin_prefetch(energy + ((size_t)b * NN + n + 32) * NN + m, 0, 1);
    float e = energy[((size_t)b * NN + n) * NN + m];
    v2f rs = rstat[b * NN + n];
    acc += __expf(e - rs.x) * rs.y;
  }
  partial[(size_t)(sub & 7) * (NB * NN) + b * NN + m] = acc;
}

__global__ __launch_bounds__(256) void k_colfin(
    const float* __restrict__ partial, float* __restrict__ invcs)
{
  int idx = blockIdx.x * 256 + threadIdx.x;
  float s = 1e-9f;
#pragma unroll
  for (int c = 0; c < 8; ++c) s += partial[(size_t)c * (NB * NN) + idx];
  invcs[idx] = 1.f / s;
}

// --- attn apply: d = h - (xv @ softmax-attn)/colsum : f16 WMMA, double-buffered LDS ---
__global__ __launch_bounds__(256) void k_attn(
    const float* __restrict__ energy, const v2f* __restrict__ rstat,
    const float* __restrict__ invcs, const _Float16* __restrict__ xv,
    const float* __restrict__ h, float* __restrict__ dbuf)
{
  __shared__ __align__(16) _Float16 pT[2 * 16 * 48];  // [buf][m_local][k], stride 48
  const int b = blockIdx.y, m0 = blockIdx.x * 16, tid = threadIdx.x;
  const int lane = tid & 31, w = tid >> 5;
  const int hl = lane >> 4, ml = lane & 15;
  const int c0 = w * 16;
  const int nl0 = tid >> 4, nl1 = nl0 + 16, mmv = tid & 15;

  v8f acc = {};
  const _Float16* xrow = xv + ((size_t)b * CC + c0 + ml) * NN + hl * 8;

  // stage 0 preload
  float ev0, ev1; v2f rs0, rs1;
  {
    int n = nl0;
    ev0 = energy[((size_t)b * NN + n) * NN + m0 + mmv]; rs0 = rstat[b * NN + n];
    n = nl1;
    ev1 = energy[((size_t)b * NN + n) * NN + m0 + mmv]; rs1 = rstat[b * NN + n];
  }

  for (int kb = 0; kb < NN; kb += 32) {
    _Float16* buf = pT + ((kb >> 5) & 1) * (16 * 48);
    buf[mmv * 48 + nl0] = (_Float16)(__expf(ev0 - rs0.x) * rs0.y);
    buf[mmv * 48 + nl1] = (_Float16)(__expf(ev1 - rs1.x) * rs1.y);
    __syncthreads();

    if (kb + 32 < NN) {           // prefetch next tile's operands behind the WMMA
      int n = kb + 32 + nl0;
      ev0 = energy[((size_t)b * NN + n) * NN + m0 + mmv]; rs0 = rstat[b * NN + n];
      n = kb + 32 + nl1;
      ev1 = energy[((size_t)b * NN + n) * NN + m0 + mmv]; rs1 = rstat[b * NN + n];
    }

    V16 a;
    const _Float16* ap = xrow + kb;
    a.q[0] = *(const u32x4*)(ap);
    a.q[1] = *(const u32x4*)(ap + 16);

    V16 bm;
    const _Float16* bp = buf + ml * 48 + hl * 8;
    bm.q[0] = *(const u32x4*)(bp);
    bm.q[1] = *(const u32x4*)(bp + 16);

    acc = __builtin_amdgcn_wmma_f32_16x16x32_f16(
        false, a.v, false, bm.v, (short)0, acc, false, false);
  }

  const int col = m0 + ml;
  const float icl = invcs[b * NN + col];
#pragma unroll
  for (int r = 0; r < 8; ++r) {
    int c = c0 + r + hl * 8;
    size_t o = ((size_t)b * CC + c) * NN + col;
    dbuf[o] = h[o] - acc[r] * icl;
  }
}

// ------------- final: out = h + relu(bn2(trans_w @ d + trans_b)) -------------
__global__ __launch_bounds__(256) void k_final(
    const float* __restrict__ h, const float* __restrict__ dbuf,
    const float* __restrict__ tw, const float* __restrict__ tb,
    const float* __restrict__ g, const float* __restrict__ be,
    const float* __restrict__ mu, const float* __restrict__ var,
    float* __restrict__ out)
{
  __shared__ float dt[CC][16];
  const int b = blockIdx.y, n0 = blockIdx.x * 16, tid = threadIdx.x;
#pragma unroll
  for (int i = 0; i < 8; ++i) {
    int e = tid + (i << 8);
    dt[e >> 4][e & 15] = dbuf[((size_t)b * CC + (e >> 4)) * NN + n0 + (e & 15)];
  }
  __syncthreads();
#pragma unroll
  for (int i = 0; i < 8; ++i) {
    int e = tid + (i << 8);
    int o = e >> 4, nl = e & 15;
    float acc = tb[o];
#pragma unroll
    for (int c = 0; c < CC; ++c) acc += tw[o * CC + c] * dt[c][nl];
    float sc = g[o] * rsqrtf(var[o] + BN_EPS);
    float vv = acc * sc + (be[o] - mu[o] * sc);
    vv = fmaxf(vv, 0.f);
    size_t off = ((size_t)b * CC + o) * NN + n0 + nl;
    out[off] = h[off] + vv;
  }
}

extern "C" void kernel_launch(void* const* d_in, const int* in_sizes, int n_in,
                              void* d_out, int out_size, void* d_ws, size_t ws_size,
                              hipStream_t stream)
{
  const float* x      = (const float*)d_in[0];
  const float* conv1w = (const float*)d_in[1];
  const float* g1 = (const float*)d_in[2];
  const float* b1 = (const float*)d_in[3];
  const float* m1 = (const float*)d_in[4];
  const float* v1 = (const float*)d_in[5];
  const float* qkw = (const float*)d_in[6];
  const float* vw  = (const float*)d_in[7];
  const float* vb  = (const float*)d_in[8];
  const float* tw  = (const float*)d_in[9];
  const float* tb  = (const float*)d_in[10];
  const float* g2 = (const float*)d_in[11];
  const float* b2 = (const float*)d_in[12];
  const float* m2 = (const float*)d_in[13];
  const float* v2 = (const float*)d_in[14];

  float* out    = (float*)d_out;
  float* energy = out + (size_t)NB * CC * NN;      // outputs: (out, energy) concat

  float* h       = (float*)d_ws;                   // 16 MB
  float* dbuf    = h    + (size_t)NB * CC * NN;    // 16 MB
  v2f*   rstat   = (v2f*)(dbuf + (size_t)NB * CC * NN);  // 256 KB {rowmax, 1/rowsum}
  float* invcs   = (float*)(rstat + NB * NN);      // 128 KB
  float* colpart = invcs + NB * NN;                // 1 MB
  float* qkT32   = colpart + 8 * NB * NN;          // 4 MB (f32, L2-resident)
  _Float16* xv   = (_Float16*)(qkT32 + (size_t)NB * NN * CQ); // 8 MB

  dim3 blk(256);
  k_stem    <<<dim3(NN / 16, NB), blk, 0, stream>>>(x, conv1w, g1, b1, m1, v1, h);
  k_qkv     <<<dim3(NN / 16, NB), blk, 0, stream>>>(h, qkw, vw, vb, qkT32, xv);
  k_energy  <<<dim3(4096),        blk, 0, stream>>>(qkT32, energy);
  k_rowstats<<<dim3(NB * NN),     blk, 0, stream>>>(energy, rstat);
  k_colpart <<<dim3(1024),        blk, 0, stream>>>(energy, rstat, colpart);
  k_colfin  <<<dim3(128),         blk, 0, stream>>>(colpart, invcs);
  k_attn    <<<dim3(NN / 16, NB), blk, 0, stream>>>(energy, rstat, invcs, xv, h, dbuf);
  k_final   <<<dim3(NN / 16, NB), blk, 0, stream>>>(h, dbuf, tw, tb, g2, b2, m2, v2, out);
}
